// NetVLAD_73538430042886
// MI455X (gfx1250) — matrix operand order repaired and checked by reading, not compiled, not run on hardware
//
#include <hip/hip_runtime.h>
#include <hip/hip_bf16.h>
#include <stdint.h>

// ---------------------------------------------------------------------------
// NetVLAD for MI455X (gfx1250): bf16 WMMA pipeline, wave32,
// async-to-LDS double-buffered B-staging in the dominant VLAD GEMM.
// ---------------------------------------------------------------------------

typedef __bf16 bf16;
typedef bf16  v16bf __attribute__((ext_vector_type(16)));
typedef float v8f   __attribute__((ext_vector_type(8)));
typedef unsigned int u32x4 __attribute__((ext_vector_type(4)));
typedef unsigned int u32x2 __attribute__((ext_vector_type(2)));

#define NIMG 32
#define CCH  128
#define HWPX 9216      // 96*96
#define KCL  64
#define KC   8192      // K*C

static __device__ __forceinline__ unsigned short f2bf(float f) {
  unsigned int u = __float_as_uint(f);
  u += 0x7fffu + ((u >> 16) & 1u);       // round-to-nearest-even
  return (unsigned short)(u >> 16);
}
static __device__ __forceinline__ unsigned int pack2(float lo, float hi) {
  return (unsigned int)f2bf(lo) | ((unsigned int)f2bf(hi) << 16);
}

union FragBF { v16bf f; unsigned int u[8]; u32x4 q[2]; };

// ---------------------------------------------------------------------------
// 1) Per-pixel L2 normalization over channels. One wave per pixel.
//    Writes xn  [n*HW + h][c]  (contiguous c: A-matrix friendly, logits GEMM)
//    Writes xnT [n][c][h]      (contiguous h: B-matrix friendly, VLAD GEMM)
// ---------------------------------------------------------------------------
__global__ void k_norm(const float* __restrict__ x,
                       unsigned short* __restrict__ xn,
                       unsigned short* __restrict__ xnT) {
  int wave = (blockIdx.x * blockDim.x + threadIdx.x) >> 5;
  int lane = threadIdx.x & 31;
  if (wave >= NIMG * HWPX) return;
  int n = wave / HWPX, hw = wave % HWPX;
  const float* xp = x + (size_t)n * CCH * HWPX + hw;
  int c0 = lane * 4;
  float v0 = xp[(size_t)(c0 + 0) * HWPX];
  float v1 = xp[(size_t)(c0 + 1) * HWPX];
  float v2 = xp[(size_t)(c0 + 2) * HWPX];
  float v3 = xp[(size_t)(c0 + 3) * HWPX];
  float s = v0 * v0 + v1 * v1 + v2 * v2 + v3 * v3;
  for (int m = 16; m >= 1; m >>= 1) s += __shfl_xor(s, m, 32);
  float inv = 1.0f / fmaxf(sqrtf(s), 1e-12f);
  v0 *= inv; v1 *= inv; v2 *= inv; v3 *= inv;
  u32x2 d; d.x = pack2(v0, v1); d.y = pack2(v2, v3);
  *(u32x2*)(xn + (size_t)wave * CCH + c0) = d;
  unsigned short* t = xnT + (size_t)n * CCH * HWPX + hw;
  t[(size_t)(c0 + 0) * HWPX] = f2bf(v0);
  t[(size_t)(c0 + 1) * HWPX] = f2bf(v1);
  t[(size_t)(c0 + 2) * HWPX] = f2bf(v2);
  t[(size_t)(c0 + 3) * HWPX] = f2bf(v3);
}

// ---------------------------------------------------------------------------
// 2) Weight conversion f32 -> bf16 (conv_w [K][C], proj_w [C][K*C])
// ---------------------------------------------------------------------------
__global__ void k_cvt(const float* __restrict__ conv_w,
                      const float* __restrict__ proj_w,
                      unsigned short* __restrict__ cw,
                      unsigned short* __restrict__ pw) {
  int total = KCL * CCH + CCH * KC;
  for (int i = blockIdx.x * blockDim.x + threadIdx.x; i < total;
       i += gridDim.x * blockDim.x) {
    if (i < KCL * CCH) cw[i] = f2bf(conv_w[i]);
    else               pw[i - KCL * CCH] = f2bf(proj_w[i - KCL * CCH]);
  }
}

// ---------------------------------------------------------------------------
// 3) Logits GEMM (16 pixels x 64 clusters, inner C=128) + fused softmax.
//    Stores soft bf16 as [n][k][h] (VLAD A-layout friendly) + ssum atomics.
// ---------------------------------------------------------------------------
__global__ void k_logits(const unsigned short* __restrict__ xn,
                         const unsigned short* __restrict__ cw,
                         unsigned short* __restrict__ soft,
                         float* __restrict__ ssum) {
  int wave = (blockIdx.x * blockDim.x + threadIdx.x) >> 5;
  int lane = threadIdx.x & 31;
  if (wave >= NIMG * (HWPX / 16)) return;
  int n  = wave / (HWPX / 16);
  int h0 = (wave % (HWPX / 16)) * 16;
  int m  = lane & 15;        // row within tile (pixel)
  int kh = lane >> 4;        // K-half select per ISA 16-bit A layout

  v8f acc[4];
  for (int t = 0; t < 4; ++t)
    for (int i = 0; i < 8; ++i) acc[t][i] = 0.0f;

  const unsigned int* arow =
      (const unsigned int*)(xn + (size_t)(n * HWPX + h0 + m) * CCH);
  for (int cb = 0; cb < 4; ++cb) {              // 32-channel K-steps
    FragBF A;
    int adw = cb * 16 + kh * 4;                 // A: K pairs {0,1}..{6,7}, {16,17}..
    A.q[0] = *(const u32x4*)(arow + adw);
    A.q[1] = *(const u32x4*)(arow + adw + 8);
    for (int kt = 0; kt < 4; ++kt) {            // 4 cluster tiles (k = 0..63)
      FragBF B;
      const unsigned int* brow =
          (const unsigned int*)(cw + (size_t)(kt * 16 + m) * CCH);
      int bdw = cb * 16 + kh * 4;               // B: K pairs {0,1}..{6,7}, {16,17}..
      B.q[0] = *(const u32x4*)(brow + bdw);
      B.q[1] = *(const u32x4*)(brow + bdw + 8);
      acc[kt] = __builtin_amdgcn_wmma_f32_16x16x32_bf16(
          false, A.f, false, B.f, (short)0, acc[kt], false, false);
    }
  }

  // Softmax over 64 clusters per pixel row; xor-shuffles {8,4,2,1} reduce
  // exactly the 16-lane half that carries one D-fragment row.
#pragma unroll
  for (int r = 0; r < 8; ++r) {
    float mx = fmaxf(fmaxf(acc[0][r], acc[1][r]), fmaxf(acc[2][r], acc[3][r]));
    for (int sh = 8; sh >= 1; sh >>= 1) mx = fmaxf(mx, __shfl_xor(mx, sh, 32));
    float e0 = __expf(acc[0][r] - mx);
    float e1 = __expf(acc[1][r] - mx);
    float e2 = __expf(acc[2][r] - mx);
    float e3 = __expf(acc[3][r] - mx);
    float s = e0 + e1 + e2 + e3;
    for (int sh = 8; sh >= 1; sh >>= 1) s += __shfl_xor(s, sh, 32);
    float inv = 1.0f / s;
    acc[0][r] = e0 * inv; acc[1][r] = e1 * inv;
    acc[2][r] = e2 * inv; acc[3][r] = e3 * inv;
  }

  for (int kt = 0; kt < 4; ++kt) {
    int k = kt * 16 + m;
    size_t base = ((size_t)(n * KCL + k)) * HWPX + h0 + kh * 8;
    u32x4 d;
    d.x = pack2(acc[kt][0], acc[kt][1]);
    d.y = pack2(acc[kt][2], acc[kt][3]);
    d.z = pack2(acc[kt][4], acc[kt][5]);
    d.w = pack2(acc[kt][6], acc[kt][7]);
    *(u32x4*)(soft + base) = d;
    float loc = acc[kt][0] + acc[kt][1] + acc[kt][2] + acc[kt][3] +
                acc[kt][4] + acc[kt][5] + acc[kt][6] + acc[kt][7];
    atomicAdd(&ssum[n * KCL + k], loc);
  }
}

// ---------------------------------------------------------------------------
// 4) VLAD GEMM: V[n][k][c] = sum_h soft[n][k][h] * xnT[n][c][h].
//    Block = 4 waves (kt = 0..3) for one (n, ct); the shared 32h x 16c
//    B-tile (1 KB) is staged into LDS via GLOBAL_LOAD_ASYNC_TO_LDS_B128
//    (ASYNCcnt), double-buffered, produced by wave 0 one step ahead.
//    A-fragments are register-prefetched one step ahead.
// ---------------------------------------------------------------------------
__global__ void k_vlad(const unsigned short* __restrict__ soft,
                       const unsigned short* __restrict__ xnT,
                       float* __restrict__ vlad) {
  __shared__ unsigned short tile[2][16 * 32];   // 2 x 1KB, rows of 64B
  const int NSTEP = HWPX / 32;                  // 288

  int tid = threadIdx.x;
  int waveId = tid >> 5, lane = tid & 31;
  int blk = blockIdx.x;                         // 256 blocks
  int n = blk >> 3, ct = blk & 7;
  int kt = waveId;
  int m = lane & 15, kh = lane >> 4;

  // --- producer (wave 0) setup: 64 x 16B chunks per tile, 2 issues/lane ---
  unsigned long long bbase =
      (unsigned long long)(const void*)(xnT + (size_t)(n * CCH + ct * 16) * HWPX);
  int prow = lane >> 2;                         // rows 0..7 (rows 8..15 via imm)
  unsigned voff = (unsigned)(prow * HWPX * 2 + (lane & 3) * 16);
  unsigned lds0 = (unsigned)(size_t)&tile[0][0];
  unsigned ldsChunk = lds0 + (unsigned)(lane * 16);

  if (waveId == 0) {                            // produce buffer 0 (h0 = 0)
    asm volatile("global_load_async_to_lds_b128 %0, %1, %2 offset:0"
                 :: "v"(ldsChunk), "v"(voff), "s"(bbase) : "memory");
    asm volatile("global_load_async_to_lds_b128 %0, %1, %2 offset:147456"
                 :: "v"(ldsChunk + 512u), "v"(voff), "s"(bbase) : "memory");
    asm volatile("s_wait_asynccnt 0" ::: "memory");
  }

  // --- consumer setup: A fragment for step 0 ---
  const unsigned int* arow =
      (const unsigned int*)(soft + ((size_t)(n * KCL + kt * 16 + m)) * HWPX);
  FragBF A;
  A.q[0] = *(const u32x4*)(arow + kh * 4);
  A.q[1] = *(const u32x4*)(arow + kh * 4 + 8);

  v8f acc;
  for (int i = 0; i < 8; ++i) acc[i] = 0.0f;

  __syncthreads();                              // buffer 0 visible to all waves

  for (int hb = 0; hb < NSTEP; ++hb) {
    int cur = hb & 1;
    // produce next B tile while this step computes
    if (waveId == 0 && (hb + 1) < NSTEP) {
      unsigned vo = voff + (unsigned)((hb + 1) * 64);
      unsigned la = lds0 + (unsigned)((hb + 1) & 1) * 1024u + (unsigned)(lane * 16);
      asm volatile("global_load_async_to_lds_b128 %0, %1, %2 offset:0"
                   :: "v"(la), "v"(vo), "s"(bbase) : "memory");
      asm volatile("global_load_async_to_lds_b128 %0, %1, %2 offset:147456"
                   :: "v"(la + 512u), "v"(vo), "s"(bbase) : "memory");
    }
    // register-prefetch next A fragment
    FragBF An;
    if ((hb + 1) < NSTEP) {
      int adw = (hb + 1) * 16 + kh * 4;
      An.q[0] = *(const u32x4*)(arow + adw);
      An.q[1] = *(const u32x4*)(arow + adw + 8);
    }
    // consume current B tile from LDS (ds_load_b128 x2)
    FragBF B;
    const unsigned short* tp = &tile[cur][m * 32 + kh * 8];
    B.q[0] = *(const u32x4*)(tp);
    B.q[1] = *(const u32x4*)(tp + 16);
    acc = __builtin_amdgcn_wmma_f32_16x16x32_bf16(
        false, A.f, false, B.f, (short)0, acc, false, false);

    if (waveId == 0 && (hb + 1) < NSTEP)
      asm volatile("s_wait_asynccnt 0" ::: "memory");
    __syncthreads();                            // publish next buffer
    if ((hb + 1) < NSTEP) A = An;
  }

#pragma unroll
  for (int r = 0; r < 8; ++r) {
    int k = kt * 16 + kh * 8 + r;
    vlad[((size_t)(n * KCL + k)) * CCH + ct * 16 + m] = acc[r];
  }
}

// ---------------------------------------------------------------------------
// 5) Centroid subtraction + intra-normalization; bf16 repack for projection;
//    accumulate global norm per image. One wave per (n,k) row.
// ---------------------------------------------------------------------------
__global__ void k_fin(const float* __restrict__ vlad,
                      const float* __restrict__ ssum,
                      const float* __restrict__ cent,
                      unsigned short* __restrict__ vn,
                      float* __restrict__ gnorm) {
  int wave = (blockIdx.x * blockDim.x + threadIdx.x) >> 5;
  int lane = threadIdx.x & 31;
  if (wave >= NIMG * KCL) return;
  int n = wave / KCL, k = wave % KCL;
  float s = ssum[wave];
  int c0 = lane * 4;
  const float* vp = vlad + (size_t)wave * CCH + c0;
  const float* cp = cent + (size_t)k * CCH + c0;
  float v0 = vp[0] - s * cp[0];
  float v1 = vp[1] - s * cp[1];
  float v2 = vp[2] - s * cp[2];
  float v3 = vp[3] - s * cp[3];
  float ss = v0 * v0 + v1 * v1 + v2 * v2 + v3 * v3;
  for (int sh = 16; sh >= 1; sh >>= 1) ss += __shfl_xor(ss, sh, 32);
  float inv = 1.0f / fmaxf(sqrtf(ss), 1e-12f);
  v0 *= inv; v1 *= inv; v2 *= inv; v3 *= inv;
  u32x2 d; d.x = pack2(v0, v1); d.y = pack2(v2, v3);
  *(u32x2*)(vn + (size_t)n * KC + k * CCH + c0) = d;
  if (lane == 0) atomicAdd(&gnorm[n], ss * inv * inv);
}

// ---------------------------------------------------------------------------
// 6) Projection: out[n][c] = (1/||v_n||) * sum_j vn[n][j]*proj_w[c][j] + b[c]
// ---------------------------------------------------------------------------
__global__ void k_proj(const unsigned short* __restrict__ vn,
                       const unsigned short* __restrict__ pw,
                       const float* __restrict__ gnorm,
                       const float* __restrict__ bias,
                       float* __restrict__ out) {
  int wave = (blockIdx.x * blockDim.x + threadIdx.x) >> 5;
  int lane = threadIdx.x & 31;
  if (wave >= 16) return;
  int mt = wave >> 3, ct = wave & 7;
  int m = lane & 15, kh = lane >> 4;
  v8f acc;
  for (int i = 0; i < 8; ++i) acc[i] = 0.0f;
  const unsigned int* arow =
      (const unsigned int*)(vn + (size_t)(mt * 16 + m) * KC);
  const unsigned int* brow =
      (const unsigned int*)(pw + (size_t)(ct * 16 + m) * KC);
  for (int jb = 0; jb < KC / 32; ++jb) {
    int j0 = jb * 32;
    FragBF A, B;
    int adw = (j0 >> 1) + kh * 4;
    A.q[0] = *(const u32x4*)(arow + adw);
    A.q[1] = *(const u32x4*)(arow + adw + 8);
    int bdw = (j0 >> 1) + kh * 4;
    B.q[0] = *(const u32x4*)(brow + bdw);
    B.q[1] = *(const u32x4*)(brow + bdw + 8);
    acc = __builtin_amdgcn_wmma_f32_16x16x32_bf16(
        false, A.f, false, B.f, (short)0, acc, false, false);
  }
  float b = bias[ct * 16 + m];
#pragma unroll
  for (int r = 0; r < 8; ++r) {
    int row = mt * 16 + kh * 8 + r;
    float g = 1.0f / fmaxf(sqrtf(gnorm[row]), 1e-12f);
    out[row * CCH + ct * 16 + m] = acc[r] * g + b;
  }
}

// ---------------------------------------------------------------------------
extern "C" void kernel_launch(void* const* d_in, const int* in_sizes, int n_in,
                              void* d_out, int out_size, void* d_ws, size_t ws_size,
                              hipStream_t stream) {
  (void)in_sizes; (void)n_in; (void)out_size; (void)ws_size;
  const float* x      = (const float*)d_in[0];
  const float* conv_w = (const float*)d_in[1];
  const float* cent   = (const float*)d_in[2];
  const float* proj_w = (const float*)d_in[3];
  const float* proj_b = (const float*)d_in[4];
  float* out = (float*)d_out;

  char* ws = (char*)d_ws;
  size_t off = 0;
  unsigned short* xn   = (unsigned short*)(ws + off); off += (size_t)NIMG * HWPX * CCH * 2;
  unsigned short* xnT  = (unsigned short*)(ws + off); off += (size_t)NIMG * HWPX * CCH * 2;
  unsigned short* soft = (unsigned short*)(ws + off); off += (size_t)NIMG * KCL * HWPX * 2;
  float*          vlad = (float*)(ws + off);          off += (size_t)NIMG * KCL * CCH * 4;
  unsigned short* vn   = (unsigned short*)(ws + off); off += (size_t)NIMG * KC * 2;
  unsigned short* cw   = (unsigned short*)(ws + off); off += (size_t)KCL * CCH * 2;
  unsigned short* pw   = (unsigned short*)(ws + off); off += (size_t)CCH * KC * 2;
  float*          ssum = (float*)(ws + off);          off += (size_t)NIMG * KCL * 4;
  float*          gnorm= (float*)(ws + off);          off += (size_t)NIMG * 4;

  // zero the atomic accumulators (ssum + gnorm are contiguous)
  hipMemsetAsync(ssum, 0, (NIMG * KCL + NIMG) * sizeof(float), stream);

  {
    int waves = NIMG * HWPX;                       // 294912 waves, exact
    k_norm<<<waves * 32 / 256, 256, 0, stream>>>(x, xn, xnT);
  }
  k_cvt<<<512, 256, 0, stream>>>(conv_w, proj_w, cw, pw);
  {
    int waves = NIMG * (HWPX / 16);                // 18432 waves, exact
    k_logits<<<waves * 32 / 128, 128, 0, stream>>>(xn, cw, soft, ssum);
  }
  k_vlad<<<NIMG * 8, 128, 0, stream>>>(soft, xnT, vlad);   // 256 blocks, 4 waves
  {
    int waves = NIMG * KCL;                        // 2048 waves, exact
    k_fin<<<waves * 32 / 256, 256, 0, stream>>>(vlad, ssum, cent, vn, gnorm);
  }
  k_proj<<<16, 32, 0, stream>>>(vn, pw, gnorm, proj_b, out);
}
